// SimpleDecoder_76450417869265
// MI455X (gfx1250) — compile-verified, matrix operand order checked by reading, not run on hardware
//
#include <hip/hip_runtime.h>
#include <math.h>

// ---------------------------------------------------------------------------
// SimpleDecoder (Informer-style) for MI455X / gfx1250.
// bf16 WMMA GEMMs end-to-end, double-buffered LDS, async global->LDS staging.
// ---------------------------------------------------------------------------

#define USE_ASYNC_LDS 1   // global_load_async_to_lds_b128 for A-tile staging

#define B_     8
#define LQ_    336
#define LENC_  2048
#define D_     1024
#define H_     16
#define DH_    64
#define DFF_   4096
#define MDEC_  (B_ * LQ_)      // 2688
#define MENC_  (B_ * LENC_)    // 16384
#define U_TOP  30              // 5*ceil(ln(336))

typedef __attribute__((ext_vector_type(16))) __bf16 v16bf;
typedef __attribute__((ext_vector_type(8)))  float  v8f;
typedef unsigned short ushort_t;

union Frag { v16bf v; unsigned int u[8]; };

__device__ __forceinline__ ushort_t f2bf(float f) {
  unsigned int u = __float_as_uint(f);
  unsigned int r = u + 0x7FFFu + ((u >> 16) & 1u);   // round-to-nearest-even
  return (ushort_t)(r >> 16);
}
__device__ __forceinline__ float bf2f(ushort_t h) {
  return __uint_as_float(((unsigned int)h) << 16);
}

// ---------------------------------------------------------------------------
// fp32 -> bf16 bulk convert (n % 1024 == 0), packed uint2 stores
// ---------------------------------------------------------------------------
__global__ __launch_bounds__(256) void cvt_bf16(const float* __restrict__ x,
                                                ushort_t* __restrict__ y)
{
  long i = ((long)blockIdx.x * 256 + threadIdx.x) * 4;
  float4 v = *(const float4*)(x + i);
  uint2 t;
  t.x = (unsigned)f2bf(v.x) | ((unsigned)f2bf(v.y) << 16);
  t.y = (unsigned)f2bf(v.z) | ((unsigned)f2bf(v.w) << 16);
  *(uint2*)(y + i) = t;
}

// ---------------------------------------------------------------------------
// Tiled GEMM: C[M,N] = act(A[M,K] @ W[K,N] + bias). A,W are bf16; f32 accum.
// BM=BN=128, BK=64, 256 threads (8 waves), double-buffered LDS.
// Requires M%128==0, N%128==0, K%64==0 (true for all call sites).
// ---------------------------------------------------------------------------
#define BM 128
#define BN 128
#define BK 64
#define ASTR 72   // padded ushort stride; keeps 16B alignment of K-octets
#define BSTR 72

__device__ __forceinline__ void stage_a(const ushort_t* __restrict__ A,
                                        ushort_t* sAbuf, uint4* areg,
                                        int tid, int bm, int K, int k0)
{
  for (int it = 0; it < 4; ++it) {
    int e = (it * 256 + tid) * 8;          // 8192 bf16 elements total
    int row = e >> 6, kk = e & 63;
    const ushort_t* gp = A + (long)(bm + row) * K + k0 + kk;
#if USE_ASYNC_LDS
    unsigned lds_off = (unsigned)(unsigned long long)(&sAbuf[row * ASTR + kk]);
    asm volatile("global_load_async_to_lds_b128 %0, %1, off"
                 :: "v"(lds_off), "v"((unsigned long long)gp) : "memory");
    (void)areg;
#else
    areg[it] = *(const uint4*)gp;
#endif
  }
}

#if !USE_ASYNC_LDS
__device__ __forceinline__ void store_a(ushort_t* sAbuf, const uint4* areg, int tid)
{
  for (int it = 0; it < 4; ++it) {
    int e = (it * 256 + tid) * 8;
    int row = e >> 6, kk = e & 63;
    *(uint4*)&sAbuf[row * ASTR + kk] = areg[it];
  }
}
#endif

__device__ __forceinline__ void load_b(const ushort_t* __restrict__ W,
                                       uint4* breg, int tid, int bn, int N, int k0)
{
  for (int it = 0; it < 4; ++it) {
    int e = (it * 256 + tid) * 8;          // 64 x 128 tile
    int kk = e >> 7, nn = e & 127;
    breg[it] = *(const uint4*)(W + (long)(k0 + kk) * N + bn + nn);
  }
}

__device__ __forceinline__ void store_b(ushort_t* sBbuf, const uint4* breg, int tid)
{
  for (int it = 0; it < 4; ++it) {
    int e = (it * 256 + tid) * 8;
    int kk = e >> 7, nn = e & 127;
    const ushort_t* s = (const ushort_t*)&breg[it];
    for (int j = 0; j < 8; ++j) sBbuf[(nn + j) * BSTR + kk] = s[j];  // transpose
  }
}

__global__ __launch_bounds__(256) void gemm_wmma_bf16(
    const ushort_t* __restrict__ A, const ushort_t* __restrict__ W,
    const float* __restrict__ bias, float* __restrict__ C,
    ushort_t* __restrict__ C16, int M, int N, int K, int relu)
{
  __shared__ ushort_t sA[2][BM * ASTR];   // 2 x 18KB
  __shared__ ushort_t sB[2][BN * BSTR];   // 2 x 18KB, transposed [n][k]

  const int tid  = threadIdx.x;
  const int lane = tid & 31;
  const int wid  = tid >> 5;
  const int wm   = wid & 3;       // 4 waves along M (32 rows each)
  const int wn   = wid >> 2;      // 2 waves along N (64 cols each)
  const int bm   = blockIdx.x * BM;
  const int bn   = blockIdx.y * BN;

  v8f acc[2][4];
  for (int i = 0; i < 2; ++i)
    for (int j = 0; j < 4; ++j)
      for (int e = 0; e < 8; ++e) acc[i][j][e] = 0.0f;

  const int r  = lane & 15;
  const int kb = (lane >> 4) << 3;   // ISA 16-bit fragment K sub-offset (0 / 8)

  uint4 breg[4];
#if !USE_ASYNC_LDS
  uint4 areg[4];
#else
  uint4* areg = nullptr;
#endif

  const int nk = K / BK;
  // prologue: stage first tile
  stage_a(A, sA[0], areg, tid, bm, K, 0);
  load_b(W, breg, tid, bn, N, 0);
#if !USE_ASYNC_LDS
  store_a(sA[0], areg, tid);
#endif
  store_b(sB[0], breg, tid);

  int cur = 0;
  for (int ks = 0; ks < nk; ++ks) {
#if USE_ASYNC_LDS
    asm volatile("s_wait_asynccnt 0x0" ::: "memory");
#endif
    __syncthreads();   // buf[cur] fully staged for all waves

    const bool more = (ks + 1) < nk;
    if (more) {        // issue next-tile global traffic before compute
      stage_a(A, sA[cur ^ 1], areg, tid, bm, K, (ks + 1) * BK);
      load_b(W, breg, tid, bn, N, (ks + 1) * BK);
    }

    // ---- compute on buf[cur]: 2 k-steps x 8 WMMAs ----
    for (int kk = 0; kk < BK; kk += 32) {
      Frag af[2];
      for (int sm = 0; sm < 2; ++sm) {
        int row = wm * 32 + sm * 16 + r;
        const unsigned int* p0 = (const unsigned int*)&sA[cur][row * ASTR + kk + kb];
        const unsigned int* p1 = (const unsigned int*)&sA[cur][row * ASTR + kk + 16 + kb];
        for (int j = 0; j < 4; ++j) { af[sm].u[j] = p0[j]; af[sm].u[4 + j] = p1[j]; }
      }
      Frag bfr[4];
      for (int sn = 0; sn < 4; ++sn) {
        int col = wn * 64 + sn * 16 + r;
        const unsigned int* p0 = (const unsigned int*)&sB[cur][col * BSTR + kk + kb];
        const unsigned int* p1 = (const unsigned int*)&sB[cur][col * BSTR + kk + 16 + kb];
        for (int j = 0; j < 4; ++j) { bfr[sn].u[j] = p0[j]; bfr[sn].u[4 + j] = p1[j]; }
      }
      for (int sm = 0; sm < 2; ++sm)
        for (int sn = 0; sn < 4; ++sn)
          acc[sm][sn] = __builtin_amdgcn_wmma_f32_16x16x32_bf16(
              false, af[sm].v, false, bfr[sn].v, (short)0, acc[sm][sn], false, false);
    }

    if (more) {        // LDS stores for next tile after compute (per-wave DS order)
#if !USE_ASYNC_LDS
      store_a(sA[cur ^ 1], areg, tid);
#endif
      store_b(sB[cur ^ 1], breg, tid);
    }
    cur ^= 1;
  }

  // ---- epilogue: C/D layout (VGPR j: M=j lanes 0-15, M=j+8 lanes 16-31) ----
  const int rr = (lane >> 4) * 8;
  const int cc = lane & 15;
  for (int sm = 0; sm < 2; ++sm)
    for (int sn = 0; sn < 4; ++sn)
      for (int j = 0; j < 8; ++j) {
        int row = bm + wm * 32 + sm * 16 + rr + j;
        int col = bn + wn * 64 + sn * 16 + cc;
        float v = acc[sm][sn][j] + bias[col];
        if (relu) v = fmaxf(v, 0.0f);
        if (C)   C[(long)row * N + col] = v;
        if (C16) C16[(long)row * N + col] = f2bf(v);
      }
}

// ---------------------------------------------------------------------------
// x = query_embed[l] + x_time[b,l,:] @ time_w + time_b  (writes f32 + bf16)
// ---------------------------------------------------------------------------
__global__ void embed_kernel(const float* __restrict__ qe,
                             const float* __restrict__ xt,
                             const float* __restrict__ tw,
                             const float* __restrict__ tb,
                             float* __restrict__ x, ushort_t* __restrict__ x16)
{
  int idx = blockIdx.x * blockDim.x + threadIdx.x;
  if (idx >= MDEC_ * D_) return;
  int m = idx >> 10, d = idx & (D_ - 1);
  int l = m % LQ_;
  const float* t = xt + (long)m * 3;
  float v = qe[(long)l * D_ + d] + t[0] * tw[d] + t[1] * tw[D_ + d] +
            t[2] * tw[2 * D_ + d] + tb[d];
  x[idx] = v;
  x16[idx] = f2bf(v);
}

// ---------------------------------------------------------------------------
// ProbAttention sparsity measure + top-u selection. One block per (b,h).
// ---------------------------------------------------------------------------
__global__ __launch_bounds__(256) void attn_meas_topk(
    const ushort_t* __restrict__ Q, const ushort_t* __restrict__ Kt,
    int Lk, int Upart, int* __restrict__ topIdx)
{
  int b = blockIdx.x >> 4, h = blockIdx.x & 15;
  __shared__ float Mv[LQ_];
  for (int l = threadIdx.x; l < LQ_; l += 256) {
    const ushort_t* q = Q + ((long)b * LQ_ + l) * D_ + h * DH_;
    float mx = -1e30f, sm = 0.0f;
    for (int s = 0; s < Upart; ++s) {
      unsigned int ix = ((unsigned)l * 1103515245u + (unsigned)s * 2654435761u + 12345u)
                        % (unsigned)Lk;
      const ushort_t* kp = Kt + ((long)b * Lk + ix) * D_ + h * DH_;
      float dot = 0.0f;
      for (int d = 0; d < DH_; ++d) dot += bf2f(q[d]) * bf2f(kp[d]);
      mx = fmaxf(mx, dot);
      sm += dot;
    }
    Mv[l] = mx - sm / (float)Lk;
  }
  __syncthreads();
  if (threadIdx.x == 0) {
    for (int ui = 0; ui < U_TOP; ++ui) {
      float best = -1e30f; int bi = 0;
      for (int l = 0; l < LQ_; ++l)
        if (Mv[l] > best) { best = Mv[l]; bi = l; }
      Mv[bi] = -1e31f;
      topIdx[blockIdx.x * 32 + ui] = bi;
    }
  }
}

// ---------------------------------------------------------------------------
// ctx = broadcast(mean_k V)  -> bf16 O    (one block per (b,h))
// ---------------------------------------------------------------------------
__global__ __launch_bounds__(256) void attn_mean_bcast(
    const ushort_t* __restrict__ V, ushort_t* __restrict__ O, int Lk)
{
  int b = blockIdx.x >> 4, h = blockIdx.x & 15;
  int d = threadIdx.x & 63, part = threadIdx.x >> 6;
  __shared__ float red[256];
  __shared__ ushort_t vm[DH_];
  float s = 0.0f;
  for (int k = part; k < Lk; k += 4)
    s += bf2f(V[((long)b * Lk + k) * D_ + h * DH_ + d]);
  red[threadIdx.x] = s;
  __syncthreads();
  if (part == 0)
    vm[d] = f2bf((red[d] + red[64 + d] + red[128 + d] + red[192 + d]) / (float)Lk);
  __syncthreads();
  for (int l = part; l < LQ_; l += 4)
    O[((long)b * LQ_ + l) * D_ + h * DH_ + d] = vm[d];
}

// ---------------------------------------------------------------------------
// Full softmax attention for one reduced query; scatter into O (bf16).
// One block per (b,h,ui).
// ---------------------------------------------------------------------------
__global__ __launch_bounds__(256) void attn_upd(
    const ushort_t* __restrict__ Q, const ushort_t* __restrict__ Kt,
    const ushort_t* __restrict__ V, const int* __restrict__ topIdx,
    ushort_t* __restrict__ O, int Lk)
{
  int bh = blockIdx.x / U_TOP, ui = blockIdx.x % U_TOP;
  int b = bh >> 4, h = bh & 15;
  int ti = topIdx[bh * 32 + ui];

  __shared__ float sc[LENC_];
  __shared__ float q[DH_];
  __shared__ float red[256];

  if (threadIdx.x < DH_)
    q[threadIdx.x] = bf2f(Q[((long)b * LQ_ + ti) * D_ + h * DH_ + threadIdx.x]);
  __syncthreads();

  float lmax = -1e30f;
  for (int k = threadIdx.x; k < Lk; k += 256) {
    const ushort_t* kp = Kt + ((long)b * Lk + k) * D_ + h * DH_;
    float dot = 0.0f;
    for (int d = 0; d < DH_; ++d) dot += q[d] * bf2f(kp[d]);
    dot *= 0.125f;                    // 1/sqrt(64)
    sc[k] = dot;
    lmax = fmaxf(lmax, dot);
  }
  red[threadIdx.x] = lmax; __syncthreads();
  for (int s = 128; s > 0; s >>= 1) {
    if (threadIdx.x < s) red[threadIdx.x] = fmaxf(red[threadIdx.x], red[threadIdx.x + s]);
    __syncthreads();
  }
  float gmax = red[0];
  __syncthreads();

  float lsum = 0.0f;
  for (int k = threadIdx.x; k < Lk; k += 256) {
    float e = __expf(sc[k] - gmax);
    sc[k] = e;
    lsum += e;
  }
  red[threadIdx.x] = lsum; __syncthreads();
  for (int s = 128; s > 0; s >>= 1) {
    if (threadIdx.x < s) red[threadIdx.x] += red[threadIdx.x + s];
    __syncthreads();
  }
  float inv = 1.0f / red[0];
  __syncthreads();

  int d = threadIdx.x & 63, part = threadIdx.x >> 6;
  float acc = 0.0f;
  for (int k = part; k < Lk; k += 4)
    acc += sc[k] * bf2f(V[((long)b * Lk + k) * D_ + h * DH_ + d]);
  red[threadIdx.x] = acc; __syncthreads();
  if (part == 0)
    O[((long)b * LQ_ + ti) * D_ + h * DH_ + d] =
        f2bf((red[d] + red[64 + d] + red[128 + d] + red[192 + d]) * inv);
}

// ---------------------------------------------------------------------------
// out = LayerNorm(x + x2) * g + b   (f32 + bf16 outputs; block per row)
// ---------------------------------------------------------------------------
__global__ __launch_bounds__(256) void add_ln(
    const float* __restrict__ x, const float* __restrict__ x2,
    const float* __restrict__ g, const float* __restrict__ bta,
    float* __restrict__ out, ushort_t* __restrict__ out16)
{
  long m = blockIdx.x;
  __shared__ float red[256];
  float vals[4];
  float s = 0.0f;
  for (int i = 0; i < 4; ++i) {
    int d = threadIdx.x + i * 256;
    float v = x[m * D_ + d] + x2[m * D_ + d];
    vals[i] = v; s += v;
  }
  red[threadIdx.x] = s; __syncthreads();
  for (int st = 128; st > 0; st >>= 1) {
    if (threadIdx.x < st) red[threadIdx.x] += red[threadIdx.x + st];
    __syncthreads();
  }
  float mean = red[0] * (1.0f / D_);
  __syncthreads();
  float vs = 0.0f;
  for (int i = 0; i < 4; ++i) { float t = vals[i] - mean; vs += t * t; }
  red[threadIdx.x] = vs; __syncthreads();
  for (int st = 128; st > 0; st >>= 1) {
    if (threadIdx.x < st) red[threadIdx.x] += red[threadIdx.x + st];
    __syncthreads();
  }
  float invstd = rsqrtf(red[0] * (1.0f / D_) + 1e-5f);
  for (int i = 0; i < 4; ++i) {
    int d = threadIdx.x + i * 256;
    float v = (vals[i] - mean) * invstd * g[d] + bta[d];
    out[m * D_ + d] = v;
    out16[m * D_ + d] = f2bf(v);
  }
}

// ---------------------------------------------------------------------------
// out[b,l] = sigmoid(h[m] . w2 + b2) + enc[b, -1, -1]   (block per row)
// ---------------------------------------------------------------------------
__global__ __launch_bounds__(256) void final_kernel(
    const float* __restrict__ hbuf, const float* __restrict__ w2,
    const float* __restrict__ b2, const float* __restrict__ enc,
    float* __restrict__ out)
{
  long m = blockIdx.x;
  int b = (int)(m / LQ_);
  __shared__ float red[256];
  float s = 0.0f;
  for (int d = threadIdx.x; d < 512; d += 256) s += hbuf[m * 512 + d] * w2[d];
  red[threadIdx.x] = s; __syncthreads();
  for (int st = 128; st > 0; st >>= 1) {
    if (threadIdx.x < st) red[threadIdx.x] += red[threadIdx.x + st];
    __syncthreads();
  }
  if (threadIdx.x == 0) {
    float z = red[0] + b2[0];
    float sig = 1.0f / (1.0f + __expf(-z));
    out[m] = sig + enc[((long)b * LENC_ + (LENC_ - 1)) * D_ + (D_ - 1)];
  }
}

// ---------------------------------------------------------------------------
// Host orchestration
// ---------------------------------------------------------------------------
extern "C" void kernel_launch(void* const* d_in, const int* in_sizes, int n_in,
                              void* d_out, int out_size, void* d_ws, size_t ws_size,
                              hipStream_t stream)
{
  // Input leaves in setup_inputs() insertion order (depth-first):
  const float* enc  = (const float*)d_in[0];   // [8,2048,1024]
  const float* xt   = (const float*)d_in[1];   // [8,336,3]
  const float* qe   = (const float*)d_in[2];   // [336,1024]
  const float* tw   = (const float*)d_in[3];   // [3,1024]
  const float* tb   = (const float*)d_in[4];
  // 5..12  self_attn q.w q.b k.w k.b v.w v.b o.w o.b
  // 13,14  norm0 g,b
  // 15..30 layer0: cross(8) n1(2) ff1(2) ff2(2) n2(2)
  // 31..46 layer1
  // 47,48  proj1 w,b ; 49,50 proj2 w,b
  float* out = (float*)d_out;

  // ---- workspace carve (~175 MB) ----
  char* base = (char*)d_ws;
  float* X     = (float*)base;            base += (long)MDEC_ * D_ * 4;
  float* X2    = (float*)base;            base += (long)MDEC_ * D_ * 4;
  float* P1    = (float*)base;            base += (long)MDEC_ * 512 * 4;
  ushort_t* X16   = (ushort_t*)base;      base += (long)MDEC_ * D_ * 2;
  ushort_t* ENC16 = (ushort_t*)base;      base += (long)MENC_ * D_ * 2;
  ushort_t* Q16   = (ushort_t*)base;      base += (long)MDEC_ * D_ * 2;
  ushort_t* K16   = (ushort_t*)base;      base += (long)MENC_ * D_ * 2;
  ushort_t* V16   = (ushort_t*)base;      base += (long)MENC_ * D_ * 2;
  ushort_t* O16   = (ushort_t*)base;      base += (long)MDEC_ * D_ * 2;
  ushort_t* H16   = (ushort_t*)base;      base += (long)MDEC_ * DFF_ * 2;
  ushort_t* W16   = (ushort_t*)base;      base += (long)D_ * DFF_ * 2;  // weight scratch (max 4M)
  int* topIdx = (int*)base;

  auto cvt = [&](const float* src, ushort_t* dst, long n) {
    cvt_bf16<<<(unsigned)(n / 1024), 256, 0, stream>>>(src, dst);
  };
  auto gemm = [&](const ushort_t* A, const ushort_t* Wb, const float* bias,
                  float* C, ushort_t* C16b, int M, int N, int K, int relu) {
    dim3 grid(M / BM, N / BN);
    gemm_wmma_bf16<<<grid, 256, 0, stream>>>(A, Wb, bias, C, C16b, M, N, K, relu);
  };
  auto wgemm = [&](int wi, const ushort_t* A, float* C, ushort_t* C16b,
                   int M, int N, int K, int relu) {
    cvt((const float*)d_in[wi], W16, (long)K * N);      // stream-ordered, safe reuse
    gemm(A, W16, (const float*)d_in[wi + 1], C, C16b, M, N, K, relu);
  };

  auto run_mha = [&](const ushort_t* xin, const ushort_t* ctxin, int Lk, int Mctx,
                     int Upart, int bas, float* outbuf) {
    wgemm(bas + 0, xin,   nullptr, Q16, MDEC_, D_, D_, 0);
    wgemm(bas + 2, ctxin, nullptr, K16, Mctx,  D_, D_, 0);
    wgemm(bas + 4, ctxin, nullptr, V16, Mctx,  D_, D_, 0);
    attn_meas_topk<<<B_ * H_, 256, 0, stream>>>(Q16, K16, Lk, Upart, topIdx);
    attn_mean_bcast<<<B_ * H_, 256, 0, stream>>>(V16, O16, Lk);
    attn_upd<<<B_ * H_ * U_TOP, 256, 0, stream>>>(Q16, K16, V16, topIdx, O16, Lk);
    wgemm(bas + 6, O16, outbuf, nullptr, MDEC_, D_, D_, 0);
  };

  // 0) bf16 shadow of encoder memory (used by both cross layers)
  cvt(enc, ENC16, (long)MENC_ * D_);

  // 1) embed
  embed_kernel<<<(MDEC_ * D_ + 255) / 256, 256, 0, stream>>>(qe, xt, tw, tb, X, X16);

  // 2) self attention + norm0
  run_mha(X16, X16, LQ_, MDEC_, 30, 5, X2);     // U_part = 5*ceil(ln 336) = 30
  add_ln<<<MDEC_, 256, 0, stream>>>(X, X2,
      (const float*)d_in[13], (const float*)d_in[14], X, X16);

  // 3) decoder layers
  for (int li = 0; li < 2; ++li) {
    int bas = 15 + li * 16;
    run_mha(X16, ENC16, LENC_, MENC_, 40, bas, X2);   // U_part = 40
    add_ln<<<MDEC_, 256, 0, stream>>>(X, X2,
        (const float*)d_in[bas + 8], (const float*)d_in[bas + 9], X, X16);
    wgemm(bas + 10, X16, nullptr, H16, MDEC_, DFF_, D_, 1);  // ff1 + relu (bf16 only)
    wgemm(bas + 12, H16, X2, nullptr, MDEC_, D_, DFF_, 0);   // ff2
    add_ln<<<MDEC_, 256, 0, stream>>>(X, X2,
        (const float*)d_in[bas + 14], (const float*)d_in[bas + 15], X, X16);
  }

  // 4) head: proj1(relu) -> proj2 -> sigmoid -> + last_scaled
  wgemm(47, X16, P1, nullptr, MDEC_, 512, D_, 1);
  final_kernel<<<MDEC_, 256, 0, stream>>>(P1, (const float*)d_in[49],
                                          (const float*)d_in[50], enc, out);
}
